// PGNN_70162585747787
// MI455X (gfx1250) — matrix operand order repaired and checked by reading, not compiled, not run on hardware
//
#include <hip/hip_runtime.h>
#include <hip/hip_bf16.h>
#include <math.h>

typedef __attribute__((ext_vector_type(16))) _Float16 v16h;
typedef __attribute__((ext_vector_type(8)))  _Float16 v8h;
typedef __attribute__((ext_vector_type(4)))  _Float16 v4h;
typedef __attribute__((ext_vector_type(8)))  float    v8f;

#define IN_C 128
#define HID  128
#define NCLS 40

// ---------------- degree count (atomic, self-loop added later) ----------------
__global__ __launch_bounds__(256) void deg_kernel(const int* __restrict__ ei,
                                                  float* __restrict__ deg, int E) {
  int i = blockIdx.x * blockDim.x + threadIdx.x;
  int stride = gridDim.x * blockDim.x;
  for (int e = i; e < E; e += stride) {
    atomicAdd(&deg[ei[E + e]], 1.0f);   // row 1 of edge_index = dst
  }
}

// deg -> rsqrt(deg + 1) in place (the +1 is the self-loop)
__global__ __launch_bounds__(256) void dinv_kernel(float* __restrict__ deg, int n) {
  int i = blockIdx.x * blockDim.x + threadIdx.x;
  if (i < n) deg[i] = rsqrtf(deg[i] + 1.0f);
}

// ---------------- one-shot f32 -> f16 conversion (vectorized 4-wide) ----------------
__global__ __launch_bounds__(256) void cvt_f16_kernel(const float* __restrict__ src,
                                                      _Float16* __restrict__ dst,
                                                      long total4) {
  long i = (long)blockIdx.x * blockDim.x + threadIdx.x;
  long stride = (long)gridDim.x * blockDim.x;
  const float4* s4 = (const float4*)src;
  v4h* d4 = (v4h*)dst;
  for (; i < total4; i += stride) {
    float4 v = s4[i];
    v4h h;
    h[0] = (_Float16)v.x; h[1] = (_Float16)v.y;
    h[2] = (_Float16)v.z; h[3] = (_Float16)v.w;
    d4[i] = h;
  }
}

// ---- pack W (128x128 f32 row-major) into per-lane WMMA-B f16 layout ----
// Wp[((ct*4 + kc)*32 + lane)*16 + i] = W[kc*32 + (lane>>4)*16 + i][ct*16 + (lane&15)]
// => in the GEMM, lane's B vector for (colTile=ct, kchunk=kc) is one contiguous 32B load.
__global__ __launch_bounds__(256) void packW_kernel(const float* __restrict__ W,
                                                    _Float16* __restrict__ Wp) {
  int idx = blockIdx.x * blockDim.x + threadIdx.x;   // 0 .. 16383
  int i    = idx & 15;
  int lane = (idx >> 4) & 31;
  int kc   = (idx >> 9) & 3;
  int ct   = idx >> 11;
  int k = kc * 32 + (lane >> 4) * 16 + i;
  int c = ct * 16 + (lane & 15);
  Wp[idx] = (_Float16)W[(size_t)k * HID + c];
}

// ---------------- H = Xh @ Wp via v_wmma_f32_16x16x32_f16 ----------------
// grid.x = n/16 row tiles (n % 16 == 0); 8 waves/block, wave w = column tile w.
__global__ __launch_bounds__(256) void gemm_wmma(const _Float16* __restrict__ Xh,
                                                 const _Float16* __restrict__ Wp,
                                                 float* __restrict__ H) {
  int rowTile = blockIdx.x * 16;
  int wave = threadIdx.x >> 5;          // 0..7 -> col tile (HID/16 == 8)
  int lane = threadIdx.x & 31;
  int half = lane >> 4;                 // which K-half this lane holds
  int l15  = lane & 15;
  const _Float16* xr = Xh + (size_t)(rowTile + l15) * IN_C;
  const v16h* wp = (const v16h*)Wp + (size_t)wave * 128 + lane;
  int col = wave * 16 + l15;            // B / C column for this lane

  v8f acc = {};
#pragma unroll
  for (int kc = 0; kc < 4; ++kc) {
    int k0 = kc * 32;
    // A operand, this code's contiguous k order (same as Wp): lanes 0-15 hold K = k0+0..15 (v0..7);
    //                     lanes 16-31 hold K = k0+16..31.
    v8h alo = *(const v8h*)(xr + k0 + half * 16);        // 16B load
    v8h ahi = *(const v8h*)(xr + k0 + half * 16 + 8);    // 16B load
    v16h a = __builtin_shufflevector(alo, ahi,
        0, 1, 2, 3, 4, 5, 6, 7, 8, 9, 10, 11, 12, 13, 14, 15);
    v16h b = wp[kc * 32];                                 // 32B contiguous load
    acc = __builtin_amdgcn_wmma_f32_16x16x32_f16(
        /*neg_a=*/false, a, /*neg_b=*/false, b,
        /*c_mod=*/(short)0, acc, /*reuse_a=*/false, /*reuse_b=*/false);
  }
  // C/D: VGPR r -> row (lane<16 ? r : r+8), col = lane&15  (unguarded: n%16==0)
  float* hp = H + (size_t)(rowTile + half * 8) * HID + col;
#pragma unroll
  for (int r = 0; r < 8; ++r) hp[(size_t)r * HID] = acc[r];
}

// ---------------- edge scatter: O[dst] += H[src] * dinv[src]*dinv[dst] ----------------
// one wave32 per edge; each lane handles a float4 (32*4 = 128 features)
__global__ __launch_bounds__(256) void scatter_kernel(const float* __restrict__ H,
                                                      const int* __restrict__ ei,
                                                      const float* __restrict__ dinv,
                                                      float* __restrict__ O,
                                                      int E, int n) {
  int wid  = (blockIdx.x * blockDim.x + threadIdx.x) >> 5;
  int lane = threadIdx.x & 31;
  int nw   = (gridDim.x * blockDim.x) >> 5;
  int total = E + n;                    // E edges + n self loops
  for (int e = wid; e < total; e += nw) {
    int src, dst;
    if (e < E) { src = ei[e]; dst = ei[E + e]; }
    else       { src = dst = e - E; }
    float nrm = dinv[src] * dinv[dst];
    float4 v = ((const float4*)(H + (size_t)src * HID))[lane];
    float* op = O + (size_t)dst * HID + lane * 4;
    atomicAdd(op + 0, v.x * nrm);
    atomicAdd(op + 1, v.y * nrm);
    atomicAdd(op + 2, v.z * nrm);
    atomicAdd(op + 3, v.w * nrm);
  }
}

// ---------------- fused head: bias+relu, max/min/mean, 384x40 linear, log_softmax ----------------
__global__ __launch_bounds__(128) void head_kernel(const float* __restrict__ O1,
                                                   const float* __restrict__ O2,
                                                   const float* __restrict__ O3,
                                                   const float* __restrict__ b1,
                                                   const float* __restrict__ b2,
                                                   const float* __restrict__ b3,
                                                   const float* __restrict__ linW,
                                                   const float* __restrict__ linb,
                                                   float* __restrict__ out, int n) {
  __shared__ float comb[3 * HID];
  __shared__ float logit[NCLS];
  __shared__ float red[2];
  int node = blockIdx.x;
  if (node >= n) return;
  int j = threadIdx.x;
  size_t base = (size_t)node * HID + j;
  float a1 = fmaxf(O1[base] + b1[j], 0.0f);
  float a2 = fmaxf(O2[base] + b2[j], 0.0f);
  float a3 = fmaxf(O3[base] + b3[j], 0.0f);
  comb[j]           = fmaxf(a1, fmaxf(a2, a3));
  comb[HID + j]     = fminf(a1, fminf(a2, a3));
  comb[2 * HID + j] = (a1 + a2 + a3) * (1.0f / 3.0f);
  __syncthreads();
  if (j < NCLS) {
    float acc = linb[j];
#pragma unroll 8
    for (int r = 0; r < 3 * HID; ++r)
      acc = fmaf(comb[r], linW[(size_t)r * NCLS + j], acc);
    logit[j] = acc;
  }
  __syncthreads();
  if (j == 0) {
    float m = -INFINITY;
    for (int c = 0; c < NCLS; ++c) m = fmaxf(m, logit[c]);
    float s = 0.0f;
    for (int c = 0; c < NCLS; ++c) s += __expf(logit[c] - m);
    red[0] = m;
    red[1] = __logf(s);
  }
  __syncthreads();
  if (j < NCLS) out[(size_t)node * NCLS + j] = logit[j] - red[0] - red[1];
}

extern "C" void kernel_launch(void* const* d_in, const int* in_sizes, int n_in,
                              void* d_out, int out_size, void* d_ws, size_t ws_size,
                              hipStream_t stream) {
  const float* x  = (const float*)d_in[0];
  const int* ei1  = (const int*)d_in[1];
  const int* ei2  = (const int*)d_in[2];
  const int* ei3  = (const int*)d_in[3];
  const float* W1 = (const float*)d_in[4];
  const float* b1 = (const float*)d_in[5];
  const float* W2 = (const float*)d_in[6];
  const float* b2 = (const float*)d_in[7];
  const float* W3 = (const float*)d_in[8];
  const float* b3 = (const float*)d_in[9];
  const float* lW = (const float*)d_in[10];
  const float* lb = (const float*)d_in[11];

  int n = in_sizes[0] / IN_C;     // 100000  (multiple of 16)
  int E = in_sizes[1] / 2;        // 1600000

  // workspace: 3*dinv[n] | H1..H3 [n*HID] f32 | O1..O3 [n*HID] f32 | Xh [n*HID] f16 | Wp1..3 [16K] f16
  float* p = (float*)d_ws;
  size_t NH = (size_t)n * HID;
  float* dinv1 = p;
  float* dinv2 = dinv1 + n;
  float* dinv3 = dinv2 + n;
  float* H1 = dinv3 + n;
  float* H2 = H1 + NH;
  float* H3 = H2 + NH;
  float* O1 = H3 + NH;
  float* O2 = O1 + NH;
  float* O3 = O2 + NH;
  _Float16* Xh  = (_Float16*)(O3 + NH);
  _Float16* Wp1 = Xh + NH;
  _Float16* Wp2 = Wp1 + 16384;
  _Float16* Wp3 = Wp2 + 16384;

  // zero accumulators every call (graph-capture safe)
  hipMemsetAsync(dinv1, 0, (size_t)3 * n * sizeof(float), stream);
  hipMemsetAsync(O1, 0, (size_t)3 * NH * sizeof(float), stream);

  const int tpb = 256;
  deg_kernel<<<2048, tpb, 0, stream>>>(ei1, dinv1, E);
  deg_kernel<<<2048, tpb, 0, stream>>>(ei2, dinv2, E);
  deg_kernel<<<2048, tpb, 0, stream>>>(ei3, dinv3, E);
  dinv_kernel<<<(3 * n + tpb - 1) / tpb, tpb, 0, stream>>>(dinv1, 3 * n);

  // one-shot precision/layout prep
  cvt_f16_kernel<<<4096, tpb, 0, stream>>>(x, Xh, (long)NH / 4);
  packW_kernel<<<64, tpb, 0, stream>>>(W1, Wp1);
  packW_kernel<<<64, tpb, 0, stream>>>(W2, Wp2);
  packW_kernel<<<64, tpb, 0, stream>>>(W3, Wp3);

  int rowTiles = n / 16;
  gemm_wmma<<<rowTiles, 256, 0, stream>>>(Xh, Wp1, H1);
  gemm_wmma<<<rowTiles, 256, 0, stream>>>(Xh, Wp2, H2);
  gemm_wmma<<<rowTiles, 256, 0, stream>>>(Xh, Wp3, H3);

  scatter_kernel<<<8192, 256, 0, stream>>>(H1, ei1, dinv1, O1, E, n);
  scatter_kernel<<<8192, 256, 0, stream>>>(H2, ei2, dinv2, O2, E, n);
  scatter_kernel<<<8192, 256, 0, stream>>>(H3, ei3, dinv3, O3, E, n);

  head_kernel<<<n, 128, 0, stream>>>(O1, O2, O3, b1, b2, b3, lW, lb,
                                     (float*)d_out, n);
}
